// GAtrustLike_26603027432204
// MI455X (gfx1250) — compile-verified, matrix operand order checked by reading, not run on hardware
//
#include <hip/hip_runtime.h>
#include <hip/hip_bf16.h>
#include <cstdint>

// ---------------------------------------------------------------------------
// CDNA5 (gfx1250) wave32 WMMA helpers  (V_WMMA_F32_16X16X32_F16)
// ---------------------------------------------------------------------------
typedef __attribute__((ext_vector_type(16))) _Float16 v16h;
typedef __attribute__((ext_vector_type(8)))  float    v8f;

union FragAB {
  v16h     h;
  uint32_t u[8];
  _Float16 e[16];
};

__device__ __forceinline__ v8f wmma_f16(v16h a, v16h b, v8f c) {
  // 8 args: (neg_a, A, neg_b, B, c_mod, C, reuse_a, reuse_b)
  return __builtin_amdgcn_wmma_f32_16x16x32_f16(false, a, false, b, (short)0, c,
                                                false, false);
}

// Load a 16x32 f16 A (or B^T) fragment from an LDS tile stored row-major
// with `stride` f16 elements per row.  Per ISA 7.12.2:
//   lane half hs = lane>>4; element e<8 -> k = k0 + hs*8 + e
//                           element e>=8 -> k = k0 + 16 + hs*8 + (e-8)
// Pairs (e, e+1) are contiguous -> 8 dword loads.
__device__ __forceinline__ v16h load_frag_rm(const _Float16* base, int row,
                                             int stride, int k0, int lane) {
  const int hs = (lane >> 4) & 1;
  FragAB f;
  const _Float16* p = base + row * stride + k0 + hs * 8;
#pragma unroll
  for (int pr = 0; pr < 8; ++pr) {
    const int kk = ((pr < 4) ? 0 : 16) + 2 * (pr & 3);
    f.u[pr] = *(const uint32_t*)(p + kk);
  }
  return f.h;
}

__device__ __forceinline__ float sigmoidf_(float x) {
  return 1.0f / (1.0f + __expf(-x));
}

// ---------------------------------------------------------------------------
// Kernel 1: h = tanh(x @ Wi + bi)      [N,128] @ [128,64] -> [N,64]
// 128 threads = 4 waves, each wave computes a 16x64 tile (16 WMMAs).
// ---------------------------------------------------------------------------
__global__ __launch_bounds__(128) void k_input(const float* __restrict__ x,
                                               const float* __restrict__ Wi,
                                               const float* __restrict__ bi,
                                               float* __restrict__ h, int N) {
  __shared__ _Float16 wt[64 * 128];      // Wi^T as f16: [n][k]
  __shared__ _Float16 xa[4][16 * 128];   // per-wave A tile [m][k]
  const int tid = threadIdx.x;

  for (int idx = tid; idx < 128 * 64; idx += 128) {      // coalesced Wi read
    const int k = idx >> 6, n = idx & 63;
    wt[n * 128 + k] = (_Float16)Wi[idx];
  }

  const int wave = tid >> 5, lane = tid & 31;
  const int hs = lane >> 4, ln = lane & 15;
  const int row0 = blockIdx.x * 64 + wave * 16;

  for (int idx = lane; idx < 16 * 128; idx += 32) {
    const int m = idx >> 7, k = idx & 127;
    int r = row0 + m; if (r >= N) r = N - 1;
    xa[wave][m * 128 + k] = (_Float16)x[(long)r * 128 + k];
  }
  __syncthreads();

  v8f acc[4] = {};
#pragma unroll
  for (int kt = 0; kt < 4; ++kt) {
    const v16h a = load_frag_rm(xa[wave], ln, 128, kt * 32, lane);
#pragma unroll
    for (int nt = 0; nt < 4; ++nt) {
      const v16h b = load_frag_rm(wt, nt * 16 + ln, 128, kt * 32, lane);
      acc[nt] = wmma_f16(a, b, acc[nt]);
    }
  }

#pragma unroll
  for (int nt = 0; nt < 4; ++nt) {
    const int col = nt * 16 + ln;
    const float bv = bi[col];
#pragma unroll
    for (int i = 0; i < 8; ++i) {
      const int r = row0 + hs * 8 + i;
      if (r < N) h[(long)r * 64 + col] = tanhf(acc[nt][i] + bv);
    }
  }
}

// ---------------------------------------------------------------------------
// Kernel 2: zero fill
// ---------------------------------------------------------------------------
__global__ void k_zero(float* __restrict__ p, long n) {
  long i = (long)blockIdx.x * blockDim.x + threadIdx.x;
  const long stride = (long)gridDim.x * blockDim.x;
  for (; i < n; i += stride) p[i] = 0.0f;
}

// ---------------------------------------------------------------------------
// Kernel 3: SpMM via gather + f32 atomic scatter.
// 256 threads = 4 groups of 64; lane = feature index (coalesced 64-wide),
// each group handles one nonzero per iteration.
// ---------------------------------------------------------------------------
__global__ __launch_bounds__(256) void k_spmm(const int* __restrict__ row,
                                              const int* __restrict__ col,
                                              const float* __restrict__ val,
                                              const float* __restrict__ h,
                                              float* __restrict__ out, int nnz) {
  const int lane = threadIdx.x & 63;
  const int grp  = threadIdx.x >> 6;
  for (long i = (long)blockIdx.x * 4 + grp; i < nnz; i += (long)gridDim.x * 4) {
    const int r = row[i], c = col[i];
    const float v = val[i];
    atomicAdd(&out[(long)r * 64 + lane], v * h[(long)c * 64 + lane]);
  }
}

// ---------------------------------------------------------------------------
// Kernel 4: one hop, fused:
//   gate = sigmoid([hp|hn] @ Wg + bg);  hm = gate*hp + (1-gate)*hn;
//   h    = tanh(hm @ Wo + bo)
// 128 threads = 4 waves, 16 rows per wave; hm round-trips through LDS f16
// to convert D-layout -> A-layout for the second WMMA GEMM.
// ---------------------------------------------------------------------------
__global__ __launch_bounds__(128) void k_hop(const float* __restrict__ hp,
                                             const float* __restrict__ hn,
                                             const float* __restrict__ Wg,
                                             const float* __restrict__ bg,
                                             const float* __restrict__ Wo,
                                             const float* __restrict__ bo,
                                             float* __restrict__ h, int N) {
  __shared__ _Float16 wgt[64 * 128];     // Wg^T f16  (16 KB)
  __shared__ _Float16 wot[64 * 64];      // Wo^T f16  ( 8 KB)
  __shared__ _Float16 ahp[4][16 * 64];   // hp tiles  ( 8 KB)
  __shared__ _Float16 ahn[4][16 * 64];   // hn tiles  ( 8 KB)
  __shared__ _Float16 amid[4][16 * 64];  // hm tiles  ( 8 KB)
  const int tid = threadIdx.x;

  for (int idx = tid; idx < 128 * 64; idx += 128) {
    const int k = idx >> 6, n = idx & 63;
    wgt[n * 128 + k] = (_Float16)Wg[idx];
  }
  for (int idx = tid; idx < 64 * 64; idx += 128) {
    const int k = idx >> 6, n = idx & 63;
    wot[n * 64 + k] = (_Float16)Wo[idx];
  }

  const int wave = tid >> 5, lane = tid & 31;
  const int hs = lane >> 4, ln = lane & 15;
  const int row0 = blockIdx.x * 64 + wave * 16;

  for (int idx = lane; idx < 16 * 64; idx += 32) {
    const int m = idx >> 6, k = idx & 63;
    int r = row0 + m; if (r >= N) r = N - 1;
    ahp[wave][idx] = (_Float16)hp[(long)r * 64 + k];
    ahn[wave][idx] = (_Float16)hn[(long)r * 64 + k];
  }
  __syncthreads();

  // ---- gate GEMM, K = 128 = [hp | hn] ----
  v8f acc[4] = {};
#pragma unroll
  for (int kt = 0; kt < 4; ++kt) {
    const _Float16* src = (kt < 2) ? ahp[wave] : ahn[wave];
    const v16h a = load_frag_rm(src, ln, 64, (kt & 1) * 32, lane);
#pragma unroll
    for (int nt = 0; nt < 4; ++nt) {
      const v16h b = load_frag_rm(wgt, nt * 16 + ln, 128, kt * 32, lane);
      acc[nt] = wmma_f16(a, b, acc[nt]);
    }
  }

  // ---- sigmoid + mix in D-layout, spill hm to LDS as f16 ----
#pragma unroll
  for (int nt = 0; nt < 4; ++nt) {
    const int col = nt * 16 + ln;
    const float bv = bg[col];
#pragma unroll
    for (int i = 0; i < 8; ++i) {
      const int m = hs * 8 + i;
      const float g = sigmoidf_(acc[nt][i] + bv);
      const float p = (float)ahp[wave][m * 64 + col];
      const float q = (float)ahn[wave][m * 64 + col];
      amid[wave][m * 64 + col] = (_Float16)(g * p + (1.0f - g) * q);
    }
  }
  __syncthreads();

  // ---- Wo GEMM, K = 64 ----
  v8f acc2[4] = {};
#pragma unroll
  for (int kt = 0; kt < 2; ++kt) {
    const v16h a = load_frag_rm(amid[wave], ln, 64, kt * 32, lane);
#pragma unroll
    for (int nt = 0; nt < 4; ++nt) {
      const v16h b = load_frag_rm(wot, nt * 16 + ln, 64, kt * 32, lane);
      acc2[nt] = wmma_f16(a, b, acc2[nt]);
    }
  }

#pragma unroll
  for (int nt = 0; nt < 4; ++nt) {
    const int col = nt * 16 + ln;
    const float bv = bo[col];
#pragma unroll
    for (int i = 0; i < 8; ++i) {
      const int r = row0 + hs * 8 + i;
      if (r < N) h[(long)r * 64 + col] = tanhf(acc2[nt][i] + bv);
    }
  }
}

// ---------------------------------------------------------------------------
// Kernel 5: edge scorer.  feat = [hu|hv||hu-hv||hu*hv] staged as an f16 tile
// in LDS during the (scalar) gather loop; the WMMA loop is then a plain
// LDS-fragment GEMM identical in structure to k_input.  |d| computed as
// fmaxf(d,-d) to avoid the packed-f32 fabs ISel bug.
// ---------------------------------------------------------------------------
__global__ __launch_bounds__(128) void k_edge(const float* __restrict__ h,
                                              const int* __restrict__ eu,
                                              const int* __restrict__ ev,
                                              const float* __restrict__ We1,
                                              const float* __restrict__ be1,
                                              const float* __restrict__ We2,
                                              const float* __restrict__ be2,
                                              float* __restrict__ out, int E) {
  __shared__ _Float16 wt[64 * 256];       // We1^T f16        (32 KB)
  __shared__ _Float16 ft[4][16 * 256];    // per-wave feat    (32 KB)
  const int tid = threadIdx.x;

  for (int idx = tid; idx < 256 * 64; idx += 128) {
    const int k = idx >> 6, n = idx & 63;
    wt[n * 256 + k] = (_Float16)We1[idx];
  }

  const int wave = tid >> 5, lane = tid & 31;
  const int hs = lane >> 4, ln = lane & 15;
  const int e0 = blockIdx.x * 64 + wave * 16;

  // Stage feature tile: for (edge m, feature j) compute all four blocks.
  for (int idx = lane; idx < 16 * 64; idx += 32) {
    const int m = idx >> 6, j = idx & 63;
    int e = e0 + m; if (e >= E) e = E - 1;
    const int u = eu[e], v = ev[e];
    const float a = h[(long)u * 64 + j];
    const float b = h[(long)v * 64 + j];
    const float d = a - b;
    _Float16* f = &ft[wave][m * 256];
    f[j]       = (_Float16)a;
    f[64 + j]  = (_Float16)b;
    f[128 + j] = (_Float16)fmaxf(d, -d);   // |a-b| without fabs pattern
    f[192 + j] = (_Float16)(a * b);
  }
  __syncthreads();

  v8f acc[4] = {};
#pragma unroll
  for (int kt = 0; kt < 8; ++kt) {
    const v16h a = load_frag_rm(ft[wave], ln, 256, kt * 32, lane);
#pragma unroll
    for (int nt = 0; nt < 4; ++nt) {
      const v16h b = load_frag_rm(wt, nt * 16 + ln, 256, kt * 32, lane);
      acc[nt] = wmma_f16(a, b, acc[nt]);
    }
  }

  // Epilogue: relu(z + be1) dotted with We2, reduced across the 16-lane half.
  float sums[8];
#pragma unroll
  for (int i = 0; i < 8; ++i) sums[i] = 0.0f;
#pragma unroll
  for (int nt = 0; nt < 4; ++nt) {
    const int col = nt * 16 + ln;
    const float b1 = be1[col];
    const float w2 = We2[col];
#pragma unroll
    for (int i = 0; i < 8; ++i) {
      float z = acc[nt][i] + b1;
      z = z > 0.0f ? z : 0.0f;
      sums[i] += z * w2;
    }
  }
#pragma unroll
  for (int off = 1; off < 16; off <<= 1) {
#pragma unroll
    for (int i = 0; i < 8; ++i) sums[i] += __shfl_xor(sums[i], off, 16);
  }
  if (ln == 0) {
    const float b2 = be2[0];
#pragma unroll
    for (int i = 0; i < 8; ++i) {
      const int e = e0 + hs * 8 + i;
      if (e < E) out[e] = sums[i] + b2;
    }
  }
}

// ---------------------------------------------------------------------------
// Launch
// ---------------------------------------------------------------------------
extern "C" void kernel_launch(void* const* d_in, const int* in_sizes, int n_in,
                              void* d_out, int out_size, void* d_ws, size_t ws_size,
                              hipStream_t stream) {
  (void)n_in; (void)out_size; (void)ws_size;
  const float* x       = (const float*)d_in[0];
  const int*   pos_row = (const int*)  d_in[1];
  const int*   pos_col = (const int*)  d_in[2];
  const float* pos_val = (const float*)d_in[3];
  const int*   neg_row = (const int*)  d_in[4];
  const int*   neg_col = (const int*)  d_in[5];
  const float* neg_val = (const float*)d_in[6];
  const int*   ei      = (const int*)  d_in[7];
  const float* Wi      = (const float*)d_in[8];
  const float* bi      = (const float*)d_in[9];
  const float* Wg      = (const float*)d_in[10];
  const float* bg      = (const float*)d_in[11];
  const float* Wo      = (const float*)d_in[12];
  const float* bo      = (const float*)d_in[13];
  const float* We1     = (const float*)d_in[14];
  const float* be1     = (const float*)d_in[15];
  const float* We2     = (const float*)d_in[16];
  const float* be2     = (const float*)d_in[17];

  const int N   = in_sizes[0] / 128;   // 100000
  const int NNZ = in_sizes[1];         // 1.6M
  const int E   = in_sizes[7] / 2;     // 1M

  float* h  = (float*)d_ws;            // [N,64]
  float* hp = h  + (size_t)N * 64;     // [N,64]
  float* hn = hp + (size_t)N * 64;     // [N,64]

  const dim3 blk(128);
  const int gRows = (N + 63) / 64;

  k_input<<<gRows, blk, 0, stream>>>(x, Wi, bi, h, N);

  for (int hop = 0; hop < 2; ++hop) {
    k_zero<<<2048, 256, 0, stream>>>(hp, (long)N * 128);  // hp & hn contiguous
    k_spmm<<<4096, 256, 0, stream>>>(pos_row, pos_col, pos_val, h, hp, NNZ);
    k_spmm<<<4096, 256, 0, stream>>>(neg_row, neg_col, neg_val, h, hn, NNZ);
    k_hop<<<gRows, blk, 0, stream>>>(hp, hn, Wg + (size_t)hop * 128 * 64,
                                     bg + (size_t)hop * 64,
                                     Wo + (size_t)hop * 64 * 64,
                                     bo + (size_t)hop * 64, h, N);
  }

  const int gEdge = (E + 63) / 64;
  k_edge<<<gEdge, blk, 0, stream>>>(h, ei, ei + E, We1, be1, We2, be2,
                                    (float*)d_out, E);
}
